// MLP_38792144618188
// MI455X (gfx1250) — matrix operand order repaired and checked by reading, not compiled, not run on hardware
//
#include <hip/hip_runtime.h>
#include <math.h>

// ---------------------------------------------------------------------------
// MI455X (gfx1250) fused weight-norm MLP + log_softmax.
// bf16x3 split-precision WMMA GEMMs (v_wmma_f32_16x16x32_bf16), f32 accum.
// Global->LDS staging via CDNA5 async-to-LDS engine (ASYNCcnt), double-buffered.
// ---------------------------------------------------------------------------

typedef __attribute__((ext_vector_type(16))) __bf16 v16bf;
typedef __attribute__((ext_vector_type(8)))  __bf16 v8bf;
typedef __attribute__((ext_vector_type(4)))  __bf16 v4bf;
typedef __attribute__((ext_vector_type(8)))  float  v8f;
typedef __attribute__((ext_vector_type(4)))  float  v4f;

#if defined(__HIP_DEVICE_COMPILE__) && defined(__gfx1250__)
#define USE_ASYNC_LDS 1
#else
#define USE_ASYNC_LDS 0
#endif

// 16-byte global -> LDS copy. On gfx1250: async DMA (no VGPR staging).
__device__ __forceinline__ void cp16_g2l(__bf16* lds, const __bf16* g)
{
#if USE_ASYNC_LDS
    unsigned           l32 = (unsigned)(unsigned long long)lds;   // LDS offset = low 32 bits of flat ptr
    unsigned long long g64 = (unsigned long long)g;
    asm volatile("global_load_async_to_lds_b128 %0, %1, off"
                 :: "v"(l32), "v"(g64) : "memory");
#else
    *(v8bf*)lds = *(const v8bf*)g;
#endif
}

__device__ __forceinline__ void wait_async(int pending)
{
#if USE_ASYNC_LDS
    if (pending) asm volatile("s_wait_asynccnt 0x8" ::: "memory");
    else         asm volatile("s_wait_asynccnt 0x0" ::: "memory");
#else
    (void)pending;
#endif
}

// ---------------------------------------------------------------------------
// Weight-norm + split: W = g[n]/||v[n]|| * v[n,:], stored as bf16 hi + lo.
// One block per output row n; K is a multiple of 1024.
// ---------------------------------------------------------------------------
__global__ __launch_bounds__(256)
void wn_split_kernel(const float* __restrict__ V, const float* __restrict__ G,
                     __bf16* __restrict__ Wh, __bf16* __restrict__ Wl, int K)
{
    __shared__ float red[256];
    const int n   = blockIdx.x;
    const int tid = threadIdx.x;
    const float* row = V + (size_t)n * K;

    float ss = 0.f;
    for (int k = tid * 4; k < K; k += 1024) {
        v4f v = *(const v4f*)(row + k);
        ss += v.x * v.x + v.y * v.y + v.z * v.z + v.w * v.w;
    }
    red[tid] = ss;
    __syncthreads();
    for (int s = 128; s > 0; s >>= 1) {
        if (tid < s) red[tid] += red[tid + s];
        __syncthreads();
    }
    const float scale = G[n] / sqrtf(red[0]);

    for (int k = tid * 4; k < K; k += 1024) {
        v4f v = *(const v4f*)(row + k);
        v4bf hv, lv;
#pragma unroll
        for (int j = 0; j < 4; ++j) {
            float w  = v[j] * scale;
            __bf16 h = (__bf16)w;
            hv[j] = h;
            lv[j] = (__bf16)(w - (float)h);
        }
        *(v4bf*)(Wh + (size_t)n * K + k) = hv;
        *(v4bf*)(Wl + (size_t)n * K + k) = lv;
    }
}

// ---------------------------------------------------------------------------
// Split f32 activations into bf16 hi + lo (elementwise). n multiple of 1024.
// ---------------------------------------------------------------------------
__global__ __launch_bounds__(256)
void split_f32_kernel(const float* __restrict__ X,
                      __bf16* __restrict__ Xh, __bf16* __restrict__ Xl)
{
    size_t i = ((size_t)blockIdx.x * 256 + threadIdx.x) * 4;
    v4f v = *(const v4f*)(X + i);
    v4bf hv, lv;
#pragma unroll
    for (int j = 0; j < 4; ++j) {
        __bf16 h = (__bf16)v[j];
        hv[j] = h;
        lv[j] = (__bf16)(v[j] - (float)h);
    }
    *(v4bf*)(Xh + i) = hv;
    *(v4bf*)(Xl + i) = lv;
}

// ---------------------------------------------------------------------------
// Tiled WMMA GEMM: C[M,N] = act(A[M,K] @ W[N,K]^T + bias)
// A, W given as bf16 (hi, lo) pairs; bf16x3 products, f32 accumulators.
// MODE 0: leaky-ReLU epilogue, output re-split to bf16 hi/lo.
// MODE 1: plain f32 output (logits).
// BM=BN=128, BK=32, 256 threads = 8 waves (4 M x 2 N), wave tile 32x64.
// LDS: double-buffered {Ah,Al,Wh,Wl} tiles staged by async-to-LDS DMA.
// M,N multiples of 128; K multiple of 32.
// ---------------------------------------------------------------------------
constexpr int BM = 128, BN = 128, BK = 32, LDT = BK + 8;   // +8 bf16 pad
constexpr int SA   = BM * LDT;                             // 5120 elems / sub-array
constexpr int SBUF = 4 * SA;                               // elems per LDS buffer
constexpr size_t SMEM_BYTES = (size_t)2 * SBUF * sizeof(__bf16);  // 81920 B

// Stage one {Ah,Al,Wh,Wl} 128x32 tile set into an LDS buffer.
// 8 async 16B ops per thread per tile set.
__device__ __forceinline__ void stage_tile(const __bf16* Ah, const __bf16* Al,
                                           const __bf16* Wh, const __bf16* Wl,
                                           int m0, int n0, int K, int k0,
                                           int t, __bf16* buf)
{
#pragma unroll
    for (int i = 0; i < 2; ++i) {
        int c  = t + 256 * i;          // 0..511
        int r  = c >> 2;               // tile row
        int cc = (c & 3) * 8;          // tile col (bf16)
        size_t ga = (size_t)(m0 + r) * K + k0 + cc;
        size_t gb = (size_t)(n0 + r) * K + k0 + cc;
        cp16_g2l(buf + 0 * SA + r * LDT + cc, Ah + ga);
        cp16_g2l(buf + 1 * SA + r * LDT + cc, Al + ga);
        cp16_g2l(buf + 2 * SA + r * LDT + cc, Wh + gb);
        cp16_g2l(buf + 3 * SA + r * LDT + cc, Wl + gb);
    }
}

template <int MODE>
__global__ __launch_bounds__(256)
void gemm_wn_kernel(const __bf16* __restrict__ Ah, const __bf16* __restrict__ Al,
                    const __bf16* __restrict__ Wh, const __bf16* __restrict__ Wl,
                    const float* __restrict__ bias,
                    __bf16* __restrict__ Oh, __bf16* __restrict__ Ol,
                    float* __restrict__ Of,
                    int M, int N, int K)
{
    extern __shared__ __align__(16) char smem_raw[];
    __bf16* sbuf0 = (__bf16*)smem_raw;
    __bf16* sbuf1 = sbuf0 + SBUF;

    const int m0   = blockIdx.x * BM;
    const int n0   = blockIdx.y * BN;
    const int t    = threadIdx.x;
    const int lane = t & 31;
    const int wv   = t >> 5;            // wave 0..7
    const int wm   = (wv & 3) * 32;     // wave M offset in block tile
    const int wn   = (wv >> 2) * 64;    // wave N offset in block tile
    const int lM   = lane & 15;         // M (A-frag) / N (B-frag) index
    const int lH   = lane >> 4;         // K-half selector

    v8f acc[2][4];
#pragma unroll
    for (int i = 0; i < 2; ++i)
#pragma unroll
        for (int j = 0; j < 4; ++j)
#pragma unroll
            for (int v = 0; v < 8; ++v) acc[i][j][v] = 0.f;

    const int NT = K / BK;
    stage_tile(Ah, Al, Wh, Wl, m0, n0, K, 0, t, sbuf0);

    for (int it = 0; it < NT; ++it) {
        __bf16* cur = (it & 1) ? sbuf1 : sbuf0;
        const int more = (it + 1 < NT);
        if (more)  // prefetch next tile into the other buffer while we compute
            stage_tile(Ah, Al, Wh, Wl, m0, n0, K, (it + 1) * BK, t,
                       (it & 1) ? sbuf0 : sbuf1);
        wait_async(more);   // async loads complete in order: keep <=8 (next tile) in flight
        __syncthreads();

        const __bf16* sAh = cur + 0 * SA;
        const __bf16* sAl = cur + 1 * SA;
        const __bf16* sBh = cur + 2 * SA;
        const __bf16* sBl = cur + 3 * SA;

        // A fragments: lane lM = row M; halves 0..7 = K lH*8.., 8..15 = K 16+lH*8..
        v16bf fah[2], fal[2], fbh[4], fbl[4];
#pragma unroll
        for (int mi = 0; mi < 2; ++mi) {
            const int row = wm + mi * 16 + lM;
            v8bf h0 = *(const v8bf*)(sAh + row * LDT + lH * 8);
            v8bf h1 = *(const v8bf*)(sAh + row * LDT + 16 + lH * 8);
            v8bf l0 = *(const v8bf*)(sAl + row * LDT + lH * 8);
            v8bf l1 = *(const v8bf*)(sAl + row * LDT + 16 + lH * 8);
            fah[mi] = __builtin_shufflevector(h0, h1, 0,1,2,3,4,5,6,7,8,9,10,11,12,13,14,15);
            fal[mi] = __builtin_shufflevector(l0, l1, 0,1,2,3,4,5,6,7,8,9,10,11,12,13,14,15);
        }
        // B fragments: lane lM = column N (= W row); halves 0..7 = K lH*8.., 8..15 = K 16+lH*8..
#pragma unroll
        for (int ni = 0; ni < 4; ++ni) {
            const int rowN = wn + ni * 16 + lM;
            v8bf h0 = *(const v8bf*)(sBh + rowN * LDT + lH * 8);
            v8bf h1 = *(const v8bf*)(sBh + rowN * LDT + 16 + lH * 8);
            v8bf l0 = *(const v8bf*)(sBl + rowN * LDT + lH * 8);
            v8bf l1 = *(const v8bf*)(sBl + rowN * LDT + 16 + lH * 8);
            fbh[ni] = __builtin_shufflevector(h0, h1, 0,1,2,3,4,5,6,7,8,9,10,11,12,13,14,15);
            fbl[ni] = __builtin_shufflevector(l0, l1, 0,1,2,3,4,5,6,7,8,9,10,11,12,13,14,15);
        }

        // bf16x3: hi*hi + hi*lo + lo*hi (lo*lo ~2^-16, dropped).
#pragma unroll
        for (int mi = 0; mi < 2; ++mi)
#pragma unroll
            for (int ni = 0; ni < 4; ++ni) {
                acc[mi][ni] = __builtin_amdgcn_wmma_f32_16x16x32_bf16(
                    false, fah[mi], false, fbh[ni], (short)0, acc[mi][ni], false, false);
                acc[mi][ni] = __builtin_amdgcn_wmma_f32_16x16x32_bf16(
                    false, fah[mi], false, fbl[ni], (short)0, acc[mi][ni], false, false);
                acc[mi][ni] = __builtin_amdgcn_wmma_f32_16x16x32_bf16(
                    false, fal[mi], false, fbh[ni], (short)0, acc[mi][ni], false, false);
            }
        __syncthreads();   // all waves done reading 'cur' before it is refilled
    }

    // Epilogue. C/D layout: element v of v8f -> M = v + 8*lH, N = lM.
#pragma unroll
    for (int mi = 0; mi < 2; ++mi)
#pragma unroll
        for (int ni = 0; ni < 4; ++ni) {
            const int gn = n0 + wn + ni * 16 + lM;
            const float bv = bias[gn];
#pragma unroll
            for (int v = 0; v < 8; ++v) {
                const int gm = m0 + wm + mi * 16 + lH * 8 + v;
                float val = acc[mi][ni][v] + bv;
                if (MODE == 0) {
                    val = val > 0.f ? val : 0.01f * val;   // leaky relu
                    __bf16 h = (__bf16)val;
                    size_t o = (size_t)gm * N + gn;
                    Oh[o] = h;
                    Ol[o] = (__bf16)(val - (float)h);
                } else {
                    Of[(size_t)gm * N + gn] = val;
                }
            }
        }
}

// ---------------------------------------------------------------------------
// In-place row-wise log_softmax: out[i] = x[i] - max - log(sum(exp(x-max))).
// One block per row.
// ---------------------------------------------------------------------------
__global__ __launch_bounds__(256)
void log_softmax_rows_kernel(float* __restrict__ out, int N)
{
    __shared__ float red[256];
    const int tid = threadIdx.x;
    float* row = out + (size_t)blockIdx.x * N;

    float mx = -3.402823466e+38f;
    for (int i = tid; i < N; i += 256) mx = fmaxf(mx, row[i]);
    red[tid] = mx;
    __syncthreads();
    for (int s = 128; s > 0; s >>= 1) {
        if (tid < s) red[tid] = fmaxf(red[tid], red[tid + s]);
        __syncthreads();
    }
    mx = red[0];
    __syncthreads();

    float sum = 0.f;
    for (int i = tid; i < N; i += 256) sum += expf(row[i] - mx);
    red[tid] = sum;
    __syncthreads();
    for (int s = 128; s > 0; s >>= 1) {
        if (tid < s) red[tid] += red[tid + s];
        __syncthreads();
    }
    const float c = mx + logf(red[0]);
    for (int i = tid; i < N; i += 256) row[i] -= c;
}

// ---------------------------------------------------------------------------
// Host launcher.
// Inputs: x, v0,g0,b0, v1,g1,b1, v2,g2,b2, v3,g3,b3 (all f32).
// Output: [2048, 32000] f32 log-probs.
// ---------------------------------------------------------------------------
extern "C" void kernel_launch(void* const* d_in, const int* in_sizes, int n_in,
                              void* d_out, int out_size, void* d_ws, size_t ws_size,
                              hipStream_t stream)
{
    (void)in_sizes; (void)n_in; (void)out_size; (void)ws_size;
    const int B = 2048, D = 2048, H = 4096, O = 32000;

    const float* x  = (const float*)d_in[0];
    const float* v0 = (const float*)d_in[1];
    const float* g0 = (const float*)d_in[2];
    const float* b0 = (const float*)d_in[3];
    const float* v1 = (const float*)d_in[4];
    const float* g1 = (const float*)d_in[5];
    const float* b1 = (const float*)d_in[6];
    const float* v2 = (const float*)d_in[7];
    const float* g2 = (const float*)d_in[8];
    const float* b2 = (const float*)d_in[9];
    const float* v3 = (const float*)d_in[10];
    const float* g3 = (const float*)d_in[11];
    const float* b3 = (const float*)d_in[12];
    float* out = (float*)d_out;

    // Workspace carve-up (256B aligned slices).
    char* w = (char*)d_ws;
    auto alloc = [&](size_t bytes) -> char* {
        char* p = w;
        w += (bytes + 255) & ~(size_t)255;
        return p;
    };
    __bf16* wh0 = (__bf16*)alloc((size_t)H * D * 2);
    __bf16* wl0 = (__bf16*)alloc((size_t)H * D * 2);
    __bf16* wh1 = (__bf16*)alloc((size_t)H * H * 2);
    __bf16* wl1 = (__bf16*)alloc((size_t)H * H * 2);
    __bf16* wh2 = (__bf16*)alloc((size_t)H * H * 2);
    __bf16* wl2 = (__bf16*)alloc((size_t)H * H * 2);
    __bf16* wh3 = (__bf16*)alloc((size_t)O * H * 2);
    __bf16* wl3 = (__bf16*)alloc((size_t)O * H * 2);
    __bf16* xh0 = (__bf16*)alloc((size_t)B * D * 2);
    __bf16* xl0 = (__bf16*)alloc((size_t)B * D * 2);
    __bf16* xh1 = (__bf16*)alloc((size_t)B * H * 2);
    __bf16* xl1 = (__bf16*)alloc((size_t)B * H * 2);
    __bf16* xh2 = (__bf16*)alloc((size_t)B * H * 2);
    __bf16* xl2 = (__bf16*)alloc((size_t)B * H * 2);
    // Layer-2 output reuses the layer-0-output buffers (no longer live).
    __bf16* xh3 = xh1;
    __bf16* xl3 = xl1;

    // 1) Weight-norm + bf16 hi/lo split of all weights.
    wn_split_kernel<<<H, 256, 0, stream>>>(v0, g0, wh0, wl0, D);
    wn_split_kernel<<<H, 256, 0, stream>>>(v1, g1, wh1, wl1, H);
    wn_split_kernel<<<H, 256, 0, stream>>>(v2, g2, wh2, wl2, H);
    wn_split_kernel<<<O, 256, 0, stream>>>(v3, g3, wh3, wl3, H);

    // 2) Split input activations.
    split_f32_kernel<<<((size_t)B * D) / 1024, 256, 0, stream>>>(x, xh0, xl0);

    // 3) Fused GEMM + bias + leaky-ReLU chain (epilogue re-splits to bf16).
    //    blockIdx.x = M tiles (16, dispatched adjacently -> weight reuse in L2).
    gemm_wn_kernel<0><<<dim3(B / 128, H / 128), 256, SMEM_BYTES, stream>>>(
        xh0, xl0, wh0, wl0, b0, xh1, xl1, nullptr, B, H, D);
    gemm_wn_kernel<0><<<dim3(B / 128, H / 128), 256, SMEM_BYTES, stream>>>(
        xh1, xl1, wh1, wl1, b1, xh2, xl2, nullptr, B, H, H);
    gemm_wn_kernel<0><<<dim3(B / 128, H / 128), 256, SMEM_BYTES, stream>>>(
        xh2, xl2, wh2, wl2, b2, xh3, xl3, nullptr, B, H, H);

    // 4) Final layer: f32 logits straight into d_out.
    gemm_wn_kernel<1><<<dim3(B / 128, O / 128), 256, SMEM_BYTES, stream>>>(
        xh3, xl3, wh3, wl3, b3, nullptr, nullptr, out, B, O, H);

    // 5) Row-wise log_softmax in place.
    log_softmax_rows_kernel<<<B, 256, 0, stream>>>(out, O);
}
